// graphSAGEmodel_19292993094303
// MI455X (gfx1250) — matrix operand order repaired and checked by reading, not compiled, not run on hardware
//
#include <hip/hip_runtime.h>

typedef _Float16 half_t;
typedef __attribute__((ext_vector_type(8)))  _Float16 v8h;
typedef __attribute__((ext_vector_type(16))) _Float16 v16h;
typedef __attribute__((ext_vector_type(8)))  float    v8f;

#define NP  100000
#define NA  50000
#define DD  128
#define CC  349
#define EPP 400000
#define EAP 200000
#define EPA 200000

// ---------------- elementwise helpers ----------------

__global__ void f32_to_f16_k(const float* __restrict__ in, half_t* __restrict__ out, int n) {
  int i = blockIdx.x * blockDim.x + threadIdx.x;
  int stride = gridDim.x * blockDim.x;
  for (; i < n; i += stride) out[i] = (half_t)in[i];
}

__global__ void zero_b32_k(unsigned int* __restrict__ p, int ndw) {
  int i = blockIdx.x * blockDim.x + threadIdx.x;
  int stride = gridDim.x * blockDim.x;
  for (; i < ndw; i += stride) p[i] = 0u;
}

// W: [128 x N] row-major f32  ->  WT: [NPAD x 128] f16 (N-major), zero padded rows
__global__ void transpose_w_k(const float* __restrict__ W, half_t* __restrict__ WT, int N) {
  int n = blockIdx.x;   // output row (column of W), 0..NPAD-1
  int k = threadIdx.x;  // 0..127
  WT[(size_t)n * DD + k] = (n < N) ? (half_t)W[(size_t)k * N + n] : (half_t)0.0f;
}

// ---------------- graph scatter (mean aggregation) ----------------
// Packed f16 atomic add: one GLOBAL_ATOMIC_PK_ADD_F16 covers two features,
// halving the L2 atomic-op count vs f32 atomics (the dominant cost here).
__device__ inline void atomic_pk_add_f16(half_t* addr, unsigned int packed) {
  asm volatile("global_atomic_pk_add_f16 %0, %1, off"
               :
               : "v"(addr), "v"(packed)
               : "memory");
}

// 4 edges per 256-thread block; 64 lanes per edge, one half2 per lane.
__global__ __launch_bounds__(256) void scatter_pk_k(
    const half_t* __restrict__ x, const int* __restrict__ src,
    const int* __restrict__ dst, half_t* __restrict__ agg,
    float* cnt, int E) {
  int e = blockIdx.x * 4 + (threadIdx.x >> 6);
  int t = threadIdx.x & 63;          // half2 index within the 128-wide row
  if (e >= E) return;
  int s = src[e];
  int d = dst[e];
  unsigned int v = *(const unsigned int*)(x + (size_t)s * DD + 2 * t);
  atomic_pk_add_f16(agg + (size_t)d * DD + 2 * t, v);
  if (cnt != nullptr && t == 0) atomicAdd(&cnt[d], 1.0f);
}

__global__ void finalize_mean_k(const half_t* __restrict__ agg, const float* __restrict__ cnt,
                                half_t* __restrict__ mean) {
  int n = blockIdx.x;
  float inv = 1.0f / fmaxf(cnt[n], 1.0f);
  mean[(size_t)n * DD + threadIdx.x] =
      (half_t)((float)agg[(size_t)n * DD + threadIdx.x] * inv);
}

// ---------------- fused multi-source WMMA GEMM ----------------
// out[M,N] = sum_s A_s[M,128](f16) @ B_s[128,N](f16, given as BT[NPAD,128]) + sum_s bias_s
struct GemmArgs {
  const half_t* A[4];
  const half_t* BT[4];
  const float*  bias[4];
  int S;
};

__global__ __launch_bounds__(256) void gemm_wmma_k(GemmArgs g, int M, int N, int ldc,
                                                   half_t* out_h, float* out_f) {
  const int wave = threadIdx.x >> 5;   // 8 waves/block, one 16-col tile each
  const int lane = threadIdx.x & 31;
  const int hi   = lane >> 4;          // lane half select
  const int lo   = lane & 15;
  const int row0 = blockIdx.x * 16;
  const int col0 = blockIdx.y * 128 + wave * 16;

  v8f acc = {};
  for (int s = 0; s < g.S; ++s) {
    // A fragment: 16x32 f16 tile; lane holds row M=lo,
    // K elems {hi*8..hi*8+7} and {16+hi*8..16+hi*8+7}  (ISA 7.12.2)
    const half_t* ap = g.A[s] + (size_t)(row0 + lo) * DD + hi * 8;
    // B fragment: 32x16; lane holds col N=lo, K elems {hi*8..hi*8+7} and {16+hi*8..16+hi*8+7}
    const half_t* bp = g.BT[s] + (size_t)(col0 + lo) * DD + hi * 8;
#pragma unroll
    for (int kb = 0; kb < DD; kb += 32) {
      v8h alo = *(const v8h*)(ap + kb);
      v8h ahi = *(const v8h*)(ap + kb + 16);
      v16h a = __builtin_shufflevector(alo, ahi,
                                       0, 1, 2, 3, 4, 5, 6, 7,
                                       8, 9, 10, 11, 12, 13, 14, 15);
      v8h blo = *(const v8h*)(bp + kb);
      v8h bhi = *(const v8h*)(bp + kb + 16);
      v16h b = __builtin_shufflevector(blo, bhi,
                                       0, 1, 2, 3, 4, 5, 6, 7,
                                       8, 9, 10, 11, 12, 13, 14, 15);
      acc = __builtin_amdgcn_wmma_f32_16x16x32_f16(
          /*neg_a=*/false, a, /*neg_b=*/false, b,
          /*c_mod=*/(short)0, acc, /*reuse_a=*/false, /*reuse_b=*/false);
    }
  }

  int col = col0 + lo;                 // C/D layout: N = lane%16
  if (col >= N) return;
  float bsum = 0.0f;
  for (int s = 0; s < g.S; ++s)
    if (g.bias[s]) bsum += g.bias[s][col];
#pragma unroll
  for (int r = 0; r < 8; ++r) {        // C/D layout: M = r + hi*8
    int row = row0 + r + hi * 8;
    float v = acc[r] + bsum;
    if (out_f) out_f[(size_t)row * ldc + col] = v;
    else       out_h[(size_t)row * ldc + col] = (half_t)v;
  }
}

// ---------------- host orchestration ----------------

extern "C" void kernel_launch(void* const* d_in, const int* in_sizes, int n_in,
                              void* d_out, int out_size, void* d_ws, size_t ws_size,
                              hipStream_t stream) {
  const float* xp     = (const float*)d_in[0];
  const float* xa     = (const float*)d_in[1];
  const int*   ei_pp  = (const int*)d_in[2];   // [2, EPP]: src then dst
  const int*   ei_ap  = (const int*)d_in[3];
  const int*   ei_pa  = (const int*)d_in[4];
  const float* Wl1_pp = (const float*)d_in[5];
  const float* Wr1_pp = (const float*)d_in[6];
  const float* b1_pp  = (const float*)d_in[7];
  const float* Wl1_ap = (const float*)d_in[8];
  const float* Wr1_ap = (const float*)d_in[9];
  const float* b1_ap  = (const float*)d_in[10];
  const float* Wl1_pa = (const float*)d_in[11];
  const float* Wr1_pa = (const float*)d_in[12];
  const float* b1_pa  = (const float*)d_in[13];
  const float* Wl2_pp = (const float*)d_in[14];
  const float* Wr2_pp = (const float*)d_in[15];
  const float* b2_pp  = (const float*)d_in[16];
  const float* Wl2_ap = (const float*)d_in[17];
  const float* Wr2_ap = (const float*)d_in[18];
  const float* b2_ap  = (const float*)d_in[19];
  // d_in[20..22] = Wl2_pa / Wr2_pa / b2_pa : unused by the reference
  const float* Wh     = (const float*)d_in[23];
  const float* bh     = (const float*)d_in[24];

  // ---- workspace carve-out (256B aligned) ----
  size_t off = 0;
  auto alloc = [&](size_t bytes) -> void* {
    void* p = (char*)d_ws + off;
    off += (bytes + 255) & ~(size_t)255;
    return p;
  };
  half_t* xp_h   = (half_t*)alloc((size_t)NP * DD * sizeof(half_t));
  half_t* xa_h   = (half_t*)alloc((size_t)NA * DD * sizeof(half_t));
  half_t* m_pp_h = (half_t*)alloc((size_t)NP * DD * sizeof(half_t));  // reused layer2
  half_t* m_ap_h = (half_t*)alloc((size_t)NP * DD * sizeof(half_t));  // reused layer2
  half_t* m_pa_h = (half_t*)alloc((size_t)NA * DD * sizeof(half_t));
  half_t* p1_h   = (half_t*)alloc((size_t)NP * DD * sizeof(half_t));
  half_t* a1_h   = (half_t*)alloc((size_t)NA * DD * sizeof(half_t));
  half_t* p2_h   = (half_t*)alloc((size_t)NP * DD * sizeof(half_t));
  half_t* agg_pp = (half_t*)alloc((size_t)NP * DD * sizeof(half_t));  // f16 agg, reused layer2
  half_t* agg_ap = (half_t*)alloc((size_t)NP * DD * sizeof(half_t));  // f16 agg, reused layer2
  half_t* agg_pa = (half_t*)alloc((size_t)NA * DD * sizeof(half_t));
  float*  cnt_pp = (float*)alloc((size_t)NP * sizeof(float));
  float*  cnt_ap = (float*)alloc((size_t)NP * sizeof(float));
  float*  cnt_pa = (float*)alloc((size_t)NA * sizeof(float));
  half_t* wtWl1pp = (half_t*)alloc((size_t)DD * DD * sizeof(half_t));
  half_t* wtWr1pp = (half_t*)alloc((size_t)DD * DD * sizeof(half_t));
  half_t* wtWl1ap = (half_t*)alloc((size_t)DD * DD * sizeof(half_t));
  half_t* wtWr1ap = (half_t*)alloc((size_t)DD * DD * sizeof(half_t));
  half_t* wtWl1pa = (half_t*)alloc((size_t)DD * DD * sizeof(half_t));
  half_t* wtWr1pa = (half_t*)alloc((size_t)DD * DD * sizeof(half_t));
  half_t* wtWl2pp = (half_t*)alloc((size_t)DD * DD * sizeof(half_t));
  half_t* wtWr2pp = (half_t*)alloc((size_t)DD * DD * sizeof(half_t));
  half_t* wtWl2ap = (half_t*)alloc((size_t)DD * DD * sizeof(half_t));
  half_t* wtWr2ap = (half_t*)alloc((size_t)DD * DD * sizeof(half_t));
  half_t* wtWh    = (half_t*)alloc((size_t)384 * DD * sizeof(half_t)); // 349 -> 384 pad

  auto cvt = [&](const float* in, half_t* out, int n) {
    int blocks = (n + 255) / 256; if (blocks > 4096) blocks = 4096;
    f32_to_f16_k<<<dim3(blocks), dim3(256), 0, stream>>>(in, out, n);
  };
  auto zero = [&](void* p, size_t ndw) {
    int blocks = (int)((ndw + 255) / 256); if (blocks > 4096) blocks = 4096;
    zero_b32_k<<<dim3(blocks), dim3(256), 0, stream>>>((unsigned int*)p, (int)ndw);
  };
  auto xpose = [&](const float* W, half_t* WT, int N, int NPAD) {
    transpose_w_k<<<dim3(NPAD), dim3(DD), 0, stream>>>(W, WT, N);
  };
  auto scatter = [&](const half_t* x, const int* ei, int E, half_t* agg, float* cnt) {
    scatter_pk_k<<<dim3((E + 3) / 4), dim3(256), 0, stream>>>(x, ei, ei + E, agg, cnt, E);
  };
  auto finalize = [&](const half_t* agg, const float* cnt, half_t* mean, int nnodes) {
    finalize_mean_k<<<dim3(nnodes), dim3(DD), 0, stream>>>(agg, cnt, mean);
  };
  auto gemm = [&](const GemmArgs& g, int M, int N, int ldc, half_t* oh, float* of) {
    dim3 grid(M / 16, (N + 127) / 128);
    gemm_wmma_k<<<grid, dim3(256), 0, stream>>>(g, M, N, ldc, oh, of);
  };

  // ---- precision conversion + weight prep ----
  cvt(xp, xp_h, NP * DD);
  cvt(xa, xa_h, NA * DD);
  xpose(Wl1_pp, wtWl1pp, DD, DD);  xpose(Wr1_pp, wtWr1pp, DD, DD);
  xpose(Wl1_ap, wtWl1ap, DD, DD);  xpose(Wr1_ap, wtWr1ap, DD, DD);
  xpose(Wl1_pa, wtWl1pa, DD, DD);  xpose(Wr1_pa, wtWr1pa, DD, DD);
  xpose(Wl2_pp, wtWl2pp, DD, DD);  xpose(Wr2_pp, wtWr2pp, DD, DD);
  xpose(Wl2_ap, wtWl2ap, DD, DD);  xpose(Wr2_ap, wtWr2ap, DD, DD);
  xpose(Wh,     wtWh,    CC, 384);

  // ---- layer 1: mean aggregation (packed-f16 atomics) ----
  zero(agg_pp, (size_t)NP * DD / 2); zero(agg_ap, (size_t)NP * DD / 2);
  zero(agg_pa, (size_t)NA * DD / 2);
  zero(cnt_pp, NP); zero(cnt_ap, NP); zero(cnt_pa, NA);
  scatter(xp_h, ei_pp, EPP, agg_pp, cnt_pp);
  scatter(xa_h, ei_ap, EAP, agg_ap, cnt_ap);
  scatter(xp_h, ei_pa, EPA, agg_pa, cnt_pa);
  finalize(agg_pp, cnt_pp, m_pp_h, NP);
  finalize(agg_ap, cnt_ap, m_ap_h, NP);
  finalize(agg_pa, cnt_pa, m_pa_h, NA);

  // ---- layer 1: fused GEMMs ----
  {  // p1 = m_pp@Wl1_pp + xp@Wr1_pp + b1_pp + m_ap@Wl1_ap + xp@Wr1_ap + b1_ap
    GemmArgs g = {};
    g.A[0] = m_pp_h; g.BT[0] = wtWl1pp; g.bias[0] = b1_pp;
    g.A[1] = xp_h;   g.BT[1] = wtWr1pp;
    g.A[2] = m_ap_h; g.BT[2] = wtWl1ap; g.bias[2] = b1_ap;
    g.A[3] = xp_h;   g.BT[3] = wtWr1ap;
    g.S = 4;
    gemm(g, NP, DD, DD, p1_h, nullptr);
  }
  {  // a1 = m_pa@Wl1_pa + xa@Wr1_pa + b1_pa
    GemmArgs g = {};
    g.A[0] = m_pa_h; g.BT[0] = wtWl1pa; g.bias[0] = b1_pa;
    g.A[1] = xa_h;   g.BT[1] = wtWr1pa;
    g.S = 2;
    gemm(g, NA, DD, DD, a1_h, nullptr);
  }

  // ---- layer 2: mean aggregation (counts reused, only sums recomputed) ----
  zero(agg_pp, (size_t)NP * DD / 2); zero(agg_ap, (size_t)NP * DD / 2);
  scatter(p1_h, ei_pp, EPP, agg_pp, nullptr);
  scatter(a1_h, ei_ap, EAP, agg_ap, nullptr);
  finalize(agg_pp, cnt_pp, m_pp_h, NP);
  finalize(agg_ap, cnt_ap, m_ap_h, NP);

  // ---- layer 2: fused GEMM ----
  {  // p2 = m_pp@Wl2_pp + p1@Wr2_pp + b2_pp + m_ap@Wl2_ap + p1@Wr2_ap + b2_ap
    GemmArgs g = {};
    g.A[0] = m_pp_h; g.BT[0] = wtWl2pp; g.bias[0] = b2_pp;
    g.A[1] = p1_h;   g.BT[1] = wtWr2pp;
    g.A[2] = m_ap_h; g.BT[2] = wtWl2ap; g.bias[2] = b2_ap;
    g.A[3] = p1_h;   g.BT[3] = wtWr2ap;
    g.S = 4;
    gemm(g, NP, DD, DD, p2_h, nullptr);
  }

  // ---- head: out = p2 @ Wh + bh  (f32 output, N=349) ----
  {
    GemmArgs g = {};
    g.A[0] = p2_h; g.BT[0] = wtWh; g.bias[0] = bh;
    g.S = 1;
    gemm(g, NP, CC, CC, nullptr, (float*)d_out);
  }
}
